// QJLSketch_58935541236211
// MI455X (gfx1250) — matrix operand order repaired and checked by reading, not compiled, not run on hardware
//
#include <hip/hip_runtime.h>
#include <hip/hip_bf16.h>
#include <math.h>

// ---------------------------------------------------------------------------
// QJL sketch scores, fused bf16-WMMA implementation for gfx1250 (wave32).
// ---------------------------------------------------------------------------

typedef __attribute__((ext_vector_type(16))) __bf16 v16bf;
typedef __attribute__((ext_vector_type(8)))  __bf16 v8bf;
typedef __attribute__((ext_vector_type(8)))  float  v8f;
typedef __attribute__((ext_vector_type(4)))  float  v4f;

static constexpr int Bc  = 4;
static constexpr int HQc = 32;
static constexpr int HKc = 8;
static constexpr int Mc  = 16;
static constexpr int Nc  = 8192;
static constexpr int Dc  = 128;
static constexpr int Sc  = 256;
static constexpr int Oc  = 8;
static constexpr int SOc = 64;
static constexpr int Gc  = HQc / HKc;        // 4
static constexpr int NTILE  = 128;           // n rows per workgroup
static constexpr int NTILES = Nc / NTILE;    // 64

// LDS layout (strides in elements, padded to dodge bank conflicts)
static constexpr int PT_STR   = Dc + 8;      // projT rows (bf16)
static constexpr int DT_STR   = Dc + 8;      // data tile rows (bf16)
static constexpr int SI_STR   = Sc + 8;      // sign_in rows (bf16)
static constexpr int SOUT_STR = SOc + 2;     // sign_out rows (f32)
static constexpr int CORR_STR = NTILE + 4;   // corr rows (f32)

static constexpr size_t OFF_PROJT = 0;
static constexpr size_t OFF_DATAT = OFF_PROJT + (size_t)Sc * PT_STR * 2;
static constexpr size_t OFF_SIGN  = OFF_DATAT + (size_t)NTILE * DT_STR * 2;
static constexpr size_t OFF_SOUT  = OFF_SIGN  + (size_t)NTILE * SI_STR * 2;
static constexpr size_t OFF_CORR  = OFF_SOUT  + (size_t)NTILE * SOUT_STR * 4;
static constexpr size_t OFF_PW    = OFF_CORR  + (size_t)(Gc*Mc) * CORR_STR * 4;
static constexpr size_t OFF_DVAL  = OFF_PW    + (size_t)NTILE * Oc * 4;
static constexpr size_t OFF_UIDX  = OFF_DVAL  + (size_t)NTILE * Oc * 4;
static constexpr size_t OFF_SCIN  = OFF_UIDX  + (size_t)NTILE * Oc * 4;
static constexpr size_t OFF_SCOUT = OFF_SCIN  + (size_t)NTILE * 4;
static constexpr size_t OFF_PSUM  = OFF_SCOUT + (size_t)NTILE * 4;
static constexpr size_t SMEM_BYTES = OFF_PSUM + (size_t)NTILE * 2 * 4;  // ~248 KB < 320 KB/WGP

__device__ __forceinline__ v16bf cat16(v8bf lo, v8bf hi) {
  return __builtin_shufflevector(lo, hi, 0,1,2,3,4,5,6,7,8,9,10,11,12,13,14,15);
}
__device__ __forceinline__ float sgnf(float v) {
  return (v > 0.f) ? 1.f : ((v < 0.f) ? -1.f : 0.f);
}

// ---------------------------------------------------------------------------
// Kernel A: sketched_q[b,hq,m,s] = sum_d q[b,hq,m,d] * proj[d,s]   (bf16 out)
// Tiny (1 MFLOP/block); plain VALU, q row cached in LDS.
// ---------------------------------------------------------------------------
__global__ void qjl_sketchq_kernel(const float* __restrict__ q,
                                   const float* __restrict__ proj,
                                   __bf16* __restrict__ sqw) {
  const int bh = blockIdx.x;                       // b*HQ + hq
  __shared__ float qs[Mc * Dc];
  for (int i = threadIdx.x; i < Mc * Dc; i += 256) qs[i] = q[(size_t)bh * Mc * Dc + i];
  __syncthreads();
  for (int o = threadIdx.x; o < Mc * Sc; o += 256) {
    const int m = o >> 8, s = o & (Sc - 1);
    float acc = 0.f;
#pragma unroll 8
    for (int d = 0; d < Dc; ++d) acc += qs[m * Dc + d] * proj[d * Sc + s];
    sqw[(size_t)bh * Mc * Sc + o] = (__bf16)acc;
  }
}

// ---------------------------------------------------------------------------
// Main fused kernel: one workgroup = one (b, hk, 128-row n-tile).
// ---------------------------------------------------------------------------
__global__ void qjl_scores_kernel(const float* __restrict__ query,
                                  const float* __restrict__ data,
                                  const float* __restrict__ proj,
                                  const int*   __restrict__ oidx,
                                  const __bf16* __restrict__ sqw,
                                  float* __restrict__ out) {
  extern __shared__ char smem[];
  __bf16* projT   = (__bf16*)(smem + OFF_PROJT);   // [Sc][PT_STR]  projT[s][d]
  __bf16* dataT   = (__bf16*)(smem + OFF_DATAT);   // [NTILE][DT_STR] = k_in (bf16)
  __bf16* signin  = (__bf16*)(smem + OFF_SIGN);    // [NTILE][SI_STR] sign_in (+-1)
  float*  signout = (float*) (smem + OFF_SOUT);    // [NTILE][SOUT_STR]
  float*  corr    = (float*) (smem + OFF_CORR);    // [64][CORR_STR]
  float*  pw      = (float*) (smem + OFF_PW);      // [NTILE][8]
  float*  dval    = (float*) (smem + OFF_DVAL);    // [NTILE][8]
  int*    uidx    = (int*)   (smem + OFF_UIDX);    // [NTILE][8] (-1 = dup)
  float*  scin    = (float*) (smem + OFF_SCIN);    // coef_in  * ||k_in||
  float*  scout   = (float*) (smem + OFF_SCOUT);   // coef_out * ||k_out||
  float*  psum    = (float*) (smem + OFF_PSUM);    // [NTILE][2] sumsq partials

  const float CIN  = 0.0048984503f;  // sqrt(pi/2)/256
  const float COUT = 0.0195938013f;  // sqrt(pi/2)/64

  const int blk = blockIdx.x;
  const int nt  = blk % NTILES;
  const int hk  = (blk / NTILES) % HKc;
  const int b   = blk / (NTILES * HKc);
  const int n0  = nt * NTILE;
  const int tid  = threadIdx.x;
  const int lane = tid & 31;
  const int wave = tid >> 5;
  const int lr   = lane & 15;
  const bool hi  = lane >= 16;

  const float* dbase = data + (((size_t)b * HKc + hk) * Nc + n0) * Dc;

  // ---- Phase 1a: load data tile fp32 -> bf16 LDS, accumulate sum of squares
  {
    const int row = tid >> 1, half = tid & 1;
    const float* src = dbase + (size_t)row * Dc + half * 64;
    __bf16* dst = dataT + row * DT_STR + half * 64;
    float ss = 0.f;
#pragma unroll
    for (int c = 0; c < 64; c += 4) {
      v4f v = *(const v4f*)(src + c);
      ss += v.x * v.x + v.y * v.y + v.z * v.z + v.w * v.w;
      dst[c] = (__bf16)v.x; dst[c + 1] = (__bf16)v.y;
      dst[c + 2] = (__bf16)v.z; dst[c + 3] = (__bf16)v.w;
    }
    psum[row * 2 + half] = ss;
  }
  // ---- Phase 1b: projT[s][d] = proj[d][s] in bf16 (proj lives in L2 across WGs)
  {
    const int s = tid;
    for (int d = 0; d < Dc; ++d) projT[s * PT_STR + d] = (__bf16)proj[d * Sc + s];
  }
  __syncthreads();

  // ---- Phase 1c: per-row outlier handling: dedup, norms, zero k_out dims in dataT
  if (tid < NTILE) {
    const int row = tid;
    const int* ip = oidx + (((size_t)b * HKc + hk) * Nc + n0 + row) * Oc;
    const float* drow = dbase + (size_t)row * Dc;
    int raw[Oc];
    float ssout = 0.f;
#pragma unroll
    for (int j = 0; j < Oc; ++j) {
      const int d = ip[j];
      bool dup = false;
      for (int k2 = 0; k2 < j; ++k2) dup |= (raw[k2] == d);
      raw[j] = d;
      uidx[row * Oc + j] = dup ? -1 : d;
      const float dv = dup ? 0.f : drow[d];
      dval[row * Oc + j] = dv;
      if (!dup) { ssout += dv * dv; dataT[row * DT_STR + d] = (__bf16)0.f; }
    }
    float nin = psum[row * 2] + psum[row * 2 + 1] - ssout;
    if (nin < 0.f) nin = 0.f;
    scin[row]  = CIN  * sqrtf(nin);
    scout[row] = COUT * sqrtf(ssout);
  }
  __syncthreads();

  // ---- Phase 2: sign_in = sign(k_in @ proj)   [128x128] @ [128x256], bf16 WMMA
  // wave w owns 16 rows of the tile; 16 s-tiles x 4 k-steps = 64 wmma/wave.
  {
    const __bf16* arow = dataT + (wave * 16 + lr) * DT_STR;
    for (int st = 0; st < 16; ++st) {
      const int s = st * 16 + lr;
      const __bf16* brow = projT + s * PT_STR;
      v8f acc = {0.f, 0.f, 0.f, 0.f, 0.f, 0.f, 0.f, 0.f};
#pragma unroll
      for (int kk = 0; kk < 4; ++kk) {
        const int a0 = kk * 32 + (hi ? 8 : 0);
        v16bf A = cat16(*(const v8bf*)(arow + a0), *(const v8bf*)(arow + a0 + 16));
        const int b0 = kk * 32 + (hi ? 8 : 0);
        v16bf Bm = cat16(*(const v8bf*)(brow + b0), *(const v8bf*)(brow + b0 + 16));
        acc = __builtin_amdgcn_wmma_f32_16x16x32_bf16(false, A, false, Bm,
                                                      (short)0, acc, false, false);
      }
      const int rbase = wave * 16 + (hi ? 8 : 0);
#pragma unroll
      for (int r = 0; r < 8; ++r)
        signin[(rbase + r) * SI_STR + s] = (__bf16)sgnf(acc[r]);
    }
  }

  // ---- Phase 3: sign_out[n][s<64] from the <=8 sparse dims (VALU, tiny)
  {
    const int n = tid >> 1, sh = (tid & 1) * 32;
    for (int s = sh; s < sh + 32; ++s) {
      float acc = 0.f;
#pragma unroll
      for (int j = 0; j < Oc; ++j) {
        const int d = uidx[n * Oc + j];
        if (d >= 0) acc += dval[n * Oc + j] * (float)projT[s * PT_STR + d];
      }
      signout[n * SOUT_STR + s] = sgnf(acc);
    }
  }
  __syncthreads();

  // ---- Phase 4a: pw[n][j] = sum_{s<64} sign_out[n][s]*proj[d_j][s]
  if (tid < NTILE) {
    const int n = tid;
#pragma unroll
    for (int j = 0; j < Oc; ++j) {
      const int d = uidx[n * Oc + j];
      float acc = 0.f;
      if (d >= 0)
        for (int s = 0; s < SOc; ++s)
          acc += signout[n * SOUT_STR + s] * (float)projT[s * PT_STR + d];
      pw[n * Oc + j] = acc;
    }
  }
  __syncthreads();

  // ---- Phase 4b: corr[m][n] = scout[n] * sum_j q[m, d_j] * pw[n][j]
  {
    for (int i = 0; i < 32; ++i) {
      const int o = i * 256 + tid;
      const int m = o >> 7, n = o & (NTILE - 1);
      const int g = m >> 4, mm = m & 15;
      const float* qrow = query + (((size_t)b * HQc + hk * Gc + g) * Mc + mm) * Dc;
      float acc = 0.f;
#pragma unroll
      for (int j = 0; j < Oc; ++j) {
        const int d = uidx[n * Oc + j];
        if (d >= 0) acc += qrow[d] * pw[n * Oc + j];
      }
      corr[m * CORR_STR + n] = acc * scout[n];
    }
  }
  __syncthreads();

  // ---- Phase 5: scores = sq(64x256) @ sign_in^T(256x128), bf16 WMMA, fused
  // scale + correction, store transposed (n-major, m contiguous) float4 x2.
  {
    const int ncol = wave * 16 + lr;             // local n handled by this lane
    const float scl = scin[ncol];
    const __bf16* brow = signin + ncol * SI_STR;
    for (int g = 0; g < Gc; ++g) {
      const int hq = hk * Gc + g;
      const __bf16* arow = sqw + (((size_t)b * HQc + hq) * Mc + lr) * Sc;
      v8f acc = {0.f, 0.f, 0.f, 0.f, 0.f, 0.f, 0.f, 0.f};
#pragma unroll
      for (int kk = 0; kk < 8; ++kk) {
        const int a0 = kk * 32 + (hi ? 8 : 0);
        v16bf A = cat16(*(const v8bf*)(arow + a0), *(const v8bf*)(arow + a0 + 16));
        const int b0 = kk * 32 + (hi ? 8 : 0);
        v16bf Bm = cat16(*(const v8bf*)(brow + b0), *(const v8bf*)(brow + b0 + 16));
        acc = __builtin_amdgcn_wmma_f32_16x16x32_bf16(false, A, false, Bm,
                                                      (short)0, acc, false, false);
      }
      const int mb = hi ? 8 : 0;
      const int nglob = n0 + ncol;
      float* op = out + (((size_t)b * HQc + hq) * Nc + nglob) * Mc + mb;
      v4f o0, o1;
#pragma unroll
      for (int r = 0; r < 4; ++r)
        o0[r] = acc[r] * scl + corr[(g * 16 + mb + r) * CORR_STR + ncol];
#pragma unroll
      for (int r = 0; r < 4; ++r)
        o1[r] = acc[4 + r] * scl + corr[(g * 16 + mb + 4 + r) * CORR_STR + ncol];
      *(v4f*)op = o0;
      *(v4f*)(op + 4) = o1;
    }
  }
}

// ---------------------------------------------------------------------------
extern "C" void kernel_launch(void* const* d_in, const int* in_sizes, int n_in,
                              void* d_out, int out_size, void* d_ws, size_t ws_size,
                              hipStream_t stream) {
  const float* query = (const float*)d_in[0];   // (B,HQ,M,D)
  const float* data  = (const float*)d_in[1];   // (B,HK,N,D)
  const float* proj  = (const float*)d_in[2];   // (D,S)
  const int*   oidx  = (const int*)  d_in[3];   // (B,HK,N,O)
  // d_in[4] = dim_outlier (=64), folded into compile-time SOc
  float* out = (float*)d_out;
  __bf16* sqw = (__bf16*)d_ws;                  // (B,HQ,M,S) bf16, 1 MB

  qjl_sketchq_kernel<<<Bc * HQc, 256, 0, stream>>>(query, proj, sqw);

  const int nblocks = Bc * HKc * NTILES;        // 2048
  qjl_scores_kernel<<<nblocks, 256, SMEM_BYTES, stream>>>(query, data, proj,
                                                          oidx, sqw, out);
  (void)in_sizes; (void)n_in; (void)out_size; (void)ws_size;
}
